// GraphCastDecoder_77068893159640
// MI455X (gfx1250) — matrix operand order repaired and checked by reading, not compiled, not run on hardware
//
#include <hip/hip_runtime.h>

// GraphCast m2g decoder for MI455X (gfx1250, wave32, WMMA bf16 / f32-acc).
// ~155 GFLOP, ~1.5 GB traffic -> memory-bound at 23.3 TB/s with bf16 WMMA.

#define LDIM 128      // latent / hidden width
#define HSTR 136      // LDS hidden row stride (bf16), conflict-free b128 reads
#define LSTR 132      // LDS pre-LN row stride (f32), conflict-free C writes
#define NMESH 40962
#define NGRID 262144
#define NEDGE 786432

typedef __attribute__((ext_vector_type(16))) __bf16        v16bf;
typedef __attribute__((ext_vector_type(8)))  float         v8f;
typedef __attribute__((ext_vector_type(4)))  unsigned int  u32x4;

union ABu { v16bf v; u32x4 q[2]; unsigned short s[16]; };

__device__ __forceinline__ unsigned short f2bf(float f) {
  unsigned int u = __builtin_bit_cast(unsigned int, f);
  u += 0x7FFFu + ((u >> 16) & 1u);          // round-to-nearest-even
  return (unsigned short)(u >> 16);
}
__device__ __forceinline__ float silu(float x) {
  return x * (1.0f / (1.0f + __expf(-x)));
}

// A-matrix tile load (contiguous k order, shared with ldB): lane m=lane&15 holds K
// [h*16, h*16+16) -> two contiguous 16B chunks.
__device__ __forceinline__ v16bf ldA(const unsigned short* p, int h) {
  ABu u;
  u.q[0] = *(const u32x4*)(p + h * 16);
  u.q[1] = *(const u32x4*)(p + h * 16 + 8);
  return u.v;
}
// B-matrix tile load from transposed [N][K] bf16 weights: lane's 16 K-values
// are contiguous (K = h*16 .. h*16+15 at column n).
__device__ __forceinline__ v16bf ldB(const unsigned short* p) {
  ABu u;
  u.q[0] = ((const u32x4*)p)[0];
  u.q[1] = ((const u32x4*)p)[1];
  return u.v;
}
// A tile sourced from f32 rows (agg / e buffers): convert in registers.
__device__ __forceinline__ v16bf cvtA(const float* p, int h) {
  ABu u;
#pragma unroll
  for (int i = 0; i < 8; ++i) {
    u.s[i]     = f2bf(p[h * 16 + i]);
    u.s[8 + i] = f2bf(p[h * 16 + 8 + i]);
  }
  return u.v;
}

#define WMMA_BF16(a, b, c) \
  __builtin_amdgcn_wmma_f32_16x16x32_bf16(false, (a), false, (b), (short)0, (c), false, false)

// Bias+SiLU, C-layout (lane=col, vgpr=row) -> LDS bf16 hidden tile.
__device__ __forceinline__ void write_hidden_silu(const v8f* acc, const float* b1,
                                                  unsigned short* hid, int lane15, int h) {
#pragma unroll
  for (int nt = 0; nt < 8; ++nt) {
    int col = nt * 16 + lane15;
    float bc = b1[col];
#pragma unroll
    for (int r = 0; r < 8; ++r)
      hid[(r + 8 * h) * HSTR + col] = f2bf(silu(acc[nt][r] + bc));
  }
}

// Second GEMM (128x128) from LDS hidden + bias -> pre-LN f32 LDS buffer.
// hid and lnbuf alias the same LDS slice; __syncthreads separates phases.
__device__ __forceinline__ void layer2_and_bias(const unsigned short* hid,
                                                const unsigned short* w2t,
                                                const float* b2, float* lnbuf,
                                                int lane15, int h) {
  v8f acc[8];
  const v8f vz = {0.f, 0.f, 0.f, 0.f, 0.f, 0.f, 0.f, 0.f};
#pragma unroll
  for (int i = 0; i < 8; ++i) acc[i] = vz;
  const unsigned short* arow = hid + lane15 * HSTR;
#pragma unroll
  for (int kt = 0; kt < 4; ++kt) {
    v16bf a = ldA(arow + kt * 32, h);
#pragma unroll
    for (int nt = 0; nt < 8; ++nt) {
      v16bf b = ldB(w2t + (nt * 16 + lane15) * 128 + kt * 32 + h * 16);
      acc[nt] = WMMA_BF16(a, b, acc[nt]);
    }
  }
  __syncthreads();  // hid reads done before lnbuf (aliased) writes
#pragma unroll
  for (int nt = 0; nt < 8; ++nt) {
    int col = nt * 16 + lane15;
    float bc = b2[col];
#pragma unroll
    for (int r = 0; r < 8; ++r)
      lnbuf[(r + 8 * h) * LSTR + col] = acc[nt][r] + bc;
  }
}

// Row stats for LayerNorm: 2 lanes per row, one shfl_xor combine.
__device__ __forceinline__ void ln_stats(const float* rowb, float& mean, float& rstd) {
  float s = 0.f, sq = 0.f;
#pragma unroll
  for (int c = 0; c < 64; ++c) { float v = rowb[c]; s += v; sq += v * v; }
  s  += __shfl_xor(s, 1);
  sq += __shfl_xor(sq, 1);
  mean = s * (1.0f / 128.0f);
  float var = sq * (1.0f / 128.0f) - mean * mean;
  rstd = rsqrtf(var + 1e-5f);
}

// ---------------- prep kernels ----------------
__global__ void zero_kernel(float* __restrict__ p, long n) {
  long i = ((long)blockIdx.x * blockDim.x + threadIdx.x) * 4;
  if (i >= n) return;
  *(float4*)(p + i) = make_float4(0.f, 0.f, 0.f, 0.f);
}
__global__ void cvt_kernel(const float* __restrict__ in, unsigned short* __restrict__ out, long n) {
  long i = ((long)blockIdx.x * blockDim.x + threadIdx.x) * 4;
  if (i >= n) return;
  const float4 v = *(const float4*)(in + i);
  *(uint2*)(out + i) = make_uint2(
      (unsigned)f2bf(v.x) | ((unsigned)f2bf(v.y) << 16),
      (unsigned)f2bf(v.z) | ((unsigned)f2bf(v.w) << 16));
}
// [K][N] f32 -> [N][K] bf16 (transposed for contiguous per-lane B loads)
__global__ void wconv_kernel(const float* __restrict__ w, unsigned short* __restrict__ out,
                             int K, int N) {
  int i = blockIdx.x * blockDim.x + threadIdx.x;
  if (i >= N * K) return;
  int n = i / K, k = i - n * K;
  out[i] = f2bf(w[(long)k * N + n]);
}

// ---------------- embed MLP: edge_attr [E,4] -> e [E,128] (f32) ----------------
__global__ void __launch_bounds__(128) embed_kernel(
    const float* __restrict__ edge_attr,
    const float* __restrict__ w1, const float* __restrict__ b1,
    const unsigned short* __restrict__ w2t, const float* __restrict__ b2,
    const float* __restrict__ g, const float* __restrict__ beta,
    float* __restrict__ e_out) {
  __shared__ float ln_s[4][16 * LSTR];
  const int lane = threadIdx.x & 31, wave = threadIdx.x >> 5;
  const int lane15 = lane & 15, h = lane >> 4;
  const long base = ((long)blockIdx.x * 4 + wave) * 16;
  float* lnbuf = ln_s[wave];
  unsigned short* hid = (unsigned short*)lnbuf;

  // layer 1: K=4 -> VALU (8 FLOPs/output, WMMA would waste 28/32 of K)
  {
    const int c0 = lane * 4;
    float wc[4][4], bb[4];
#pragma unroll
    for (int c = 0; c < 4; ++c) {
      bb[c] = b1[c0 + c];
#pragma unroll
      for (int k = 0; k < 4; ++k) wc[k][c] = w1[k * LDIM + c0 + c];
    }
    for (int m = 0; m < 16; ++m) {
      const float4 x = *(const float4*)(edge_attr + (base + m) * 4);
      float v[4];
#pragma unroll
      for (int c = 0; c < 4; ++c)
        v[c] = silu(bb[c] + x.x * wc[0][c] + x.y * wc[1][c] + x.z * wc[2][c] + x.w * wc[3][c]);
      *(uint2*)(hid + m * HSTR + c0) = make_uint2(
          (unsigned)f2bf(v[0]) | ((unsigned)f2bf(v[1]) << 16),
          (unsigned)f2bf(v[2]) | ((unsigned)f2bf(v[3]) << 16));
    }
  }
  __syncthreads();
  layer2_and_bias(hid, w2t, b2, lnbuf, lane15, h);
  __syncthreads();

  const int m = lane >> 1, p = lane & 1;
  const float* rowb = lnbuf + m * LSTR + p * 64;
  float mean, rstd;
  ln_stats(rowb, mean, rstd);
  float* orow = e_out + (base + m) * LDIM + p * 64;
#pragma unroll 4
  for (int c = 0; c < 64; c += 4) {
    int col = p * 64 + c;
    float4 y;
    y.x = (rowb[c + 0] - mean) * rstd * g[col + 0] + beta[col + 0];
    y.y = (rowb[c + 1] - mean) * rstd * g[col + 1] + beta[col + 1];
    y.z = (rowb[c + 2] - mean) * rstd * g[col + 2] + beta[col + 2];
    y.w = (rowb[c + 3] - mean) * rstd * g[col + 3] + beta[col + 3];
    *(float4*)(orow + c) = y;
  }
}

// ------- edge MLP: concat(mesh[src], grid[dst], e) [E,384] -> upd_e; scatter -------
// e_io holds e on entry (row owned exclusively by this wave) and e+upd_e on exit.
__global__ void __launch_bounds__(128) edge_kernel(
    const unsigned short* __restrict__ mesh_bf,
    const unsigned short* __restrict__ grid_bf,
    float* e_io,
    const int* __restrict__ esrc, const int* __restrict__ edst,
    const unsigned short* __restrict__ w1t, const float* __restrict__ b1,
    const unsigned short* __restrict__ w2t, const float* __restrict__ b2,
    const float* __restrict__ g, const float* __restrict__ beta,
    float* __restrict__ agg) {
  __shared__ float ln_s[4][16 * LSTR];
  const int lane = threadIdx.x & 31, wave = threadIdx.x >> 5;
  const int lane15 = lane & 15, h = lane >> 4;
  const long base = ((long)blockIdx.x * 4 + wave) * 16;
  float* lnbuf = ln_s[wave];
  unsigned short* hid = (unsigned short*)lnbuf;

  const long srow = esrc[base + lane15];
  const long drow = edst[base + lane15];
  const unsigned short* apM = mesh_bf + srow * LDIM;
  const unsigned short* apG = grid_bf + drow * LDIM;
  const float* apE = e_io + (base + lane15) * LDIM;

  v8f acc[8];
  const v8f vz = {0.f, 0.f, 0.f, 0.f, 0.f, 0.f, 0.f, 0.f};
#pragma unroll
  for (int i = 0; i < 8; ++i) acc[i] = vz;
#pragma unroll
  for (int kt = 0; kt < 12; ++kt) {  // K=384 = [snd | rcv | e]
    v16bf a = (kt < 4) ? ldA(apM + kt * 32, h)
            : (kt < 8) ? ldA(apG + (kt - 4) * 32, h)
                       : cvtA(apE + (kt - 8) * 32, h);
#pragma unroll
    for (int nt = 0; nt < 8; ++nt) {
      v16bf b = ldB(w1t + (nt * 16 + lane15) * 384 + kt * 32 + h * 16);
      acc[nt] = WMMA_BF16(a, b, acc[nt]);
    }
  }
  write_hidden_silu(acc, b1, hid, lane15, h);
  __syncthreads();
  layer2_and_bias(hid, w2t, b2, lnbuf, lane15, h);
  __syncthreads();

  const int m = lane >> 1, p = lane & 1;
  const float* rowb = lnbuf + m * LSTR + p * 64;
  float mean, rstd;
  ln_stats(rowb, mean, rstd);
  const long row = base + m;
  float* eio = e_io + row * LDIM + p * 64;
  float* arow = agg + (long)edst[row] * LDIM + p * 64;
  for (int c = 0; c < 64; ++c) {
    int col = p * 64 + c;
    float y = (rowb[c] - mean) * rstd * g[col] + beta[col];  // upd_e
    eio[c] = eio[c] + y;                                     // e + upd_e (read then write)
    atomicAdd(arow + c, y);                                  // segment_sum
  }
}

// ------- node MLP: concat(grid_x, agg) [N,256] -> grid_x + upd_x -------
__global__ void __launch_bounds__(128) node_kernel(
    const float* __restrict__ grid_x,
    const unsigned short* __restrict__ grid_bf,
    const float* __restrict__ agg,
    const unsigned short* __restrict__ w1t, const float* __restrict__ b1,
    const unsigned short* __restrict__ w2t, const float* __restrict__ b2,
    const float* __restrict__ g, const float* __restrict__ beta,
    float* __restrict__ out_x) {
  __shared__ float ln_s[4][16 * LSTR];
  const int lane = threadIdx.x & 31, wave = threadIdx.x >> 5;
  const int lane15 = lane & 15, h = lane >> 4;
  const long base = ((long)blockIdx.x * 4 + wave) * 16;
  float* lnbuf = ln_s[wave];
  unsigned short* hid = (unsigned short*)lnbuf;

  const unsigned short* apG = grid_bf + (base + lane15) * LDIM;
  const float* apA = agg + (base + lane15) * LDIM;

  v8f acc[8];
  const v8f vz = {0.f, 0.f, 0.f, 0.f, 0.f, 0.f, 0.f, 0.f};
#pragma unroll
  for (int i = 0; i < 8; ++i) acc[i] = vz;
#pragma unroll
  for (int kt = 0; kt < 8; ++kt) {  // K=256 = [grid_x | agg]
    v16bf a = (kt < 4) ? ldA(apG + kt * 32, h) : cvtA(apA + (kt - 4) * 32, h);
#pragma unroll
    for (int nt = 0; nt < 8; ++nt) {
      v16bf b = ldB(w1t + (nt * 16 + lane15) * 256 + kt * 32 + h * 16);
      acc[nt] = WMMA_BF16(a, b, acc[nt]);
    }
  }
  write_hidden_silu(acc, b1, hid, lane15, h);
  __syncthreads();
  layer2_and_bias(hid, w2t, b2, lnbuf, lane15, h);
  __syncthreads();

  const int m = lane >> 1, p = lane & 1;
  const float* rowb = lnbuf + m * LSTR + p * 64;
  float mean, rstd;
  ln_stats(rowb, mean, rstd);
  const long row = base + m;
  const float* gx = grid_x + row * LDIM + p * 64;
  float* orow = out_x + row * LDIM + p * 64;
#pragma unroll 4
  for (int c = 0; c < 64; ++c) {
    int col = p * 64 + c;
    float y = (rowb[c] - mean) * rstd * g[col] + beta[col];
    orow[c] = gx[c] + y;  // residual (exact f32 grid_x)
  }
}

extern "C" void kernel_launch(void* const* d_in, const int* in_sizes, int n_in,
                              void* d_out, int out_size, void* d_ws, size_t ws_size,
                              hipStream_t stream) {
  (void)in_sizes; (void)n_in; (void)out_size; (void)ws_size;
  const float* mesh_x    = (const float*)d_in[0];
  const float* grid_x    = (const float*)d_in[1];
  const float* edge_attr = (const float*)d_in[2];
  const int*   esrc      = (const int*)d_in[3];
  const int*   edst      = (const int*)d_in[4];
  const float* em_w1 = (const float*)d_in[5],  *em_b1 = (const float*)d_in[6];
  const float* em_w2 = (const float*)d_in[7],  *em_b2 = (const float*)d_in[8];
  const float* em_g  = (const float*)d_in[9],  *em_be = (const float*)d_in[10];
  const float* ed_w1 = (const float*)d_in[11], *ed_b1 = (const float*)d_in[12];
  const float* ed_w2 = (const float*)d_in[13], *ed_b2 = (const float*)d_in[14];
  const float* ed_g  = (const float*)d_in[15], *ed_be = (const float*)d_in[16];
  const float* nd_w1 = (const float*)d_in[17], *nd_b1 = (const float*)d_in[18];
  const float* nd_w2 = (const float*)d_in[19], *nd_b2 = (const float*)d_in[20];
  const float* nd_g  = (const float*)d_in[21], *nd_be = (const float*)d_in[22];

  char* ws = (char*)d_ws;
  size_t off = 0;
  auto take = [&](size_t bytes) -> char* {
    char* p = ws + off;
    off = (off + bytes + 255) & ~(size_t)255;
    return p;
  };
  unsigned short* mesh_bf = (unsigned short*)take((size_t)NMESH * LDIM * 2);
  unsigned short* grid_bf = (unsigned short*)take((size_t)NGRID * LDIM * 2);
  float*          agg     = (float*)take((size_t)NGRID * LDIM * 4);
  unsigned short* w_em2t  = (unsigned short*)take(128 * 128 * 2);
  unsigned short* w_ed1t  = (unsigned short*)take(384 * 128 * 2);
  unsigned short* w_ed2t  = (unsigned short*)take(128 * 128 * 2);
  unsigned short* w_nd1t  = (unsigned short*)take(256 * 128 * 2);
  unsigned short* w_nd2t  = (unsigned short*)take(128 * 128 * 2);

  float* out_grid = (float*)d_out;                       // [NGRID,128]
  float* out_edge = out_grid + (size_t)NGRID * LDIM;     // [NEDGE,128], holds e then e+upd_e

  const long nAgg = (long)NGRID * LDIM;
  zero_kernel<<<(int)((nAgg / 4 + 255) / 256), 256, 0, stream>>>(agg, nAgg);
  wconv_kernel<<<(128 * 128 + 255) / 256, 256, 0, stream>>>(em_w2, w_em2t, 128, 128);
  wconv_kernel<<<(384 * 128 + 255) / 256, 256, 0, stream>>>(ed_w1, w_ed1t, 384, 128);
  wconv_kernel<<<(128 * 128 + 255) / 256, 256, 0, stream>>>(ed_w2, w_ed2t, 128, 128);
  wconv_kernel<<<(256 * 128 + 255) / 256, 256, 0, stream>>>(nd_w1, w_nd1t, 256, 128);
  wconv_kernel<<<(128 * 128 + 255) / 256, 256, 0, stream>>>(nd_w2, w_nd2t, 128, 128);
  const long nMesh = (long)NMESH * LDIM, nGrid = (long)NGRID * LDIM;
  cvt_kernel<<<(int)((nMesh / 4 + 255) / 256), 256, 0, stream>>>(mesh_x, mesh_bf, nMesh);
  cvt_kernel<<<(int)((nGrid / 4 + 255) / 256), 256, 0, stream>>>(grid_x, grid_bf, nGrid);

  // 4 waves/block, 16 rows/wave -> 64 rows/block (NEDGE, NGRID divide exactly)
  embed_kernel<<<NEDGE / 64, 128, 0, stream>>>(edge_attr, em_w1, em_b1, w_em2t, em_b2,
                                               em_g, em_be, out_edge);
  edge_kernel<<<NEDGE / 64, 128, 0, stream>>>(mesh_bf, grid_bf, out_edge, esrc, edst,
                                              w_ed1t, ed_b1, w_ed2t, ed_b2, ed_g, ed_be, agg);
  node_kernel<<<NGRID / 64, 128, 0, stream>>>(grid_x, grid_bf, agg, w_nd1t, nd_b1,
                                              w_nd2t, nd_b2, nd_g, nd_be, out_grid);
}
